// HMC_LSTM_91130616086566
// MI455X (gfx1250) — matrix operand order repaired and checked by reading, not compiled, not run on hardware
//
#include <hip/hip_runtime.h>
#include <hip/hip_bf16.h>

// ---------------------------------------------------------------------------
// HMC-LSTM for MI455X (gfx1250): bf16 WMMA + async global->LDS double buffer.
// ---------------------------------------------------------------------------

#define NROWS 16384
#define KFEAT 512
#define UNITS 512
#define KX    1024              // KFEAT + UNITS
#define NCSUM 960

typedef __attribute__((ext_vector_type(16))) __bf16        v16bf;
typedef __attribute__((ext_vector_type(8)))  float         v8f;
typedef __attribute__((ext_vector_type(8)))  unsigned int  v8u;

// ----------------------------- helpers -------------------------------------

__device__ __forceinline__ unsigned short f2bf(float f) {
    unsigned u = __builtin_bit_cast(unsigned, f);
    unsigned rnd = 0x7FFFu + ((u >> 16) & 1u);   // round-to-nearest-even
    return (unsigned short)((u + rnd) >> 16);
}

// Branch-free sigmoid/tanh: v_exp_f32 + v_rcp_f32, no EXEC divergence.
__device__ __forceinline__ float fast_sigmoid(float x) {
    x = fminf(fmaxf(x, -30.0f), 30.0f);
    return __builtin_amdgcn_rcpf(1.0f + __expf(-x));
}
__device__ __forceinline__ float fast_tanh(float x) {
    x = fminf(fmaxf(x, -15.0f), 15.0f);
    float e = __expf(-2.0f * x);
    return (1.0f - e) * __builtin_amdgcn_rcpf(1.0f + e);
}

__device__ __forceinline__ v16bf as_bf16x16(uint4 lo, uint4 hi) {
    v8u t;
    t[0] = lo.x; t[1] = lo.y; t[2] = lo.z; t[3] = lo.w;
    t[4] = hi.x; t[5] = hi.y; t[6] = hi.z; t[7] = hi.w;
    return __builtin_bit_cast(v16bf, t);
}

__device__ __forceinline__ v8f wmma_bf16(v16bf a, v16bf b, v8f c) {
    return __builtin_amdgcn_wmma_f32_16x16x32_bf16(
        /*neg_a=*/false, a, /*neg_b=*/false, b,
        /*c_mod=*/(short)0, c, /*reuse_a=*/false, /*reuse_b=*/false);
}

// Async stage 32 bytes (this thread's slice) from global into LDS.
// INST_OFFSET is added to BOTH the LDS and global address (ISA 08 §4.4),
// so one address pair covers both b128 transfers.
__device__ __forceinline__ void stage_async32(const unsigned short* gsrc,
                                              unsigned short* lds) {
    unsigned l0 = (unsigned)(size_t)lds;   // low 32 bits == LDS byte offset
    asm volatile("global_load_async_to_lds_b128 %0, %1, off\n\t"
                 "global_load_async_to_lds_b128 %0, %1, off offset:32"
                 :: "v"(l0), "v"(gsrc)
                 : "memory");
}

#define WAIT_ASYNC0() asm volatile("s_wait_asynccnt 0x0" ::: "memory")

// ------------------------- prep kernels ------------------------------------

__global__ void k_f32_to_bf16(const float* __restrict__ src,
                              unsigned short* __restrict__ dst, int n) {
    int i = blockIdx.x * 256 + threadIdx.x;
    if (i < n) dst[i] = f2bf(src[i]);
}

// Pack a row-major f32 weight [Kd x Nsrc] into B fragments in this file's contiguous k order.
// Tile (inGlobal, ik) of 32x16 stored as 256 dwords: dword q = lane*8 + d,
// lane l holds column n = l&15, K base (l>>4)*16; dword d = K pair
// (kbase+2d even in [15:0], odd in [31:16]).
__global__ void k_pack_w(const float* __restrict__ W, unsigned* __restrict__ dst,
                         int Kd, int Nsrc, int coTiles) {
    int tid = blockIdx.x * 256 + threadIdx.x;
    int total = (Kd * Nsrc) >> 1;
    if (tid >= total) return;
    int ktiles = Kd >> 5;
    int t   = tid >> 8;
    int q   = tid & 255;
    int ik  = t % ktiles;
    int inS = t / ktiles;
    int lane = q >> 3;
    int d    = q & 7;
    int n  = inS * 16 + (lane & 15);
    int k0 = ik * 32 + (lane >> 4) * 16 + 2 * d;
    unsigned lo = f2bf(W[(size_t)k0 * Nsrc + n]);
    unsigned hi = f2bf(W[(size_t)(k0 + 1) * Nsrc + n]);
    dst[((size_t)(coTiles + inS) * ktiles + ik) * 256 + q] = lo | (hi << 16);
}

// --------------------- fused LSTM cell layer -------------------------------
// Grid: (NROWS/128, UNITS/64). Block: 256 threads (8 waves, 2 M x 4 N).
// Each wave: 64 rows x 16 unit-cols x 4 gates = 16 accumulators.
// Wp = packed gate weights [1024 x 2048] bf16 (f|i|o|c), ktiles=32.

#define LDS_STRIDE 40   // bf16 elements per row (32 + 8 pad)

__global__ __launch_bounds__(256)
void k_lstm_cell(const unsigned short* __restrict__ Abf,
                 const unsigned short* __restrict__ Hin,
                 unsigned short* __restrict__ Hout,
                 float* __restrict__ cand,
                 const unsigned* __restrict__ Wp,
                 const float* __restrict__ bF, const float* __restrict__ bI,
                 const float* __restrict__ bO, const float* __restrict__ bC) {
    __shared__ unsigned short As[2][128 * LDS_STRIDE];

    const int tid  = threadIdx.x;
    const int wave = tid >> 5;
    const int lane = tid & 31;
    const int waveM = wave >> 2;   // 0..1
    const int waveN = wave & 3;    // 0..3
    const int rowBlock = blockIdx.x * 128;
    const int ucolBase = blockIdx.y * 64 + waveN * 16;

    v8f acc[4][4];
#pragma unroll
    for (int g = 0; g < 4; ++g)
#pragma unroll
        for (int mi = 0; mi < 4; ++mi) acc[g][mi] = (v8f)0.0f;

    const int ldr    = tid >> 1;          // LDS row this thread fills
    const int ldhalf = tid & 1;           // which 16-bf16 half
    const size_t gOff = (size_t)(rowBlock + ldr) * 512 + ldhalf * 16;
    const int ldsOff = ldr * LDS_STRIDE + ldhalf * 16;

    // prologue: prefetch K-tile 0 into buffer 0
    stage_async32(Abf + gOff, &As[0][ldsOff]);

    for (int kk = 0; kk < 32; ++kk) {
        const int cur = kk & 1;
        WAIT_ASYNC0();          // this wave's tile landed
        __syncthreads();        // all waves' tiles landed; prev-buffer reads done

        if (kk + 1 < 32) {      // prefetch next K-tile into other buffer
            const int k1 = (kk + 1) * 32;
            const unsigned short* src =
                (k1 < KFEAT) ? (Abf + k1) : (Hin + (k1 - KFEAT));
            stage_async32(src + gOff, &As[1 - cur][ldsOff]);
        }

        // ---- B fragments: one 32x16 tile per gate, from packed L2 ----
        v16bf bfrag[4];
#pragma unroll
        for (int g = 0; g < 4; ++g) {
            size_t toff = ((size_t)(g * 32 + blockIdx.y * 4 + waveN) * 32 + kk) * 256;
            v8u b = *(const v8u*)(Wp + toff + (size_t)lane * 8);
            bfrag[g] = __builtin_bit_cast(v16bf, b);
        }

        // ---- A fragments + WMMA ----
#pragma unroll
        for (int mi = 0; mi < 4; ++mi) {
            int rloc = waveM * 64 + mi * 16 + (lane & 15);
            int koff = (lane >> 4) * 16;  // bf16 units
            const uint4* ap = (const uint4*)(&As[cur][rloc * LDS_STRIDE + koff]);
            uint4 lo = ap[0];
            uint4 hi = ap[1];             // +8 bf16 = +1 uint4
            v16bf afrag = as_bf16x16(lo, hi);
#pragma unroll
            for (int g = 0; g < 4; ++g)
                acc[g][mi] = wmma_bf16(afrag, bfrag[g], acc[g][mi]);
        }
    }

    // ---- epilogue: gates + cell update (branch-free transcendentals) ----
    const int ucol = ucolBase + (lane & 15);
    const float Bf = bF[ucol], Bi = bI[ucol], Bo = bO[ucol], Bc = bC[ucol];

#pragma unroll
    for (int mi = 0; mi < 4; ++mi) {
        int row0 = rowBlock + waveM * 64 + mi * 16 + (lane >> 4) * 8;
#pragma unroll
        for (int e = 0; e < 8; ++e) {
            size_t idx = (size_t)(row0 + e) * UNITS + ucol;
            float cprev = cand[idx];
            float fp = acc[0][mi][e] + Bf;
            float ip = acc[1][mi][e] + Bi;
            float op = acc[2][mi][e] + Bo;
            float cp = acc[3][mi][e] + Bc;
            float cn = fast_tanh(cp) * fast_sigmoid(ip) + fast_sigmoid(fp) * cprev;
            float h  = fast_sigmoid(op) * fast_tanh(cn);
            cand[idx] = cn;
            Hout[idx] = f2bf(h);
        }
    }
}

// ------------------------- logits GEMM -------------------------------------
// C[row, colOff + col] = bias[col] + sum_k A[row,k] * W[k,col]
// A = [A1 | A2] (bf16, each 512 cols wide); Kd = 512 or 1024.
// Grid: (NROWS/128, Nd/64). Block 256 = 8 waves (2 M x 4 N); wave = 64x16.

__global__ __launch_bounds__(256)
void k_gemm_logits(const unsigned short* __restrict__ A1,
                   const unsigned short* __restrict__ A2, int Kd,
                   const unsigned* __restrict__ Wp,
                   const float* __restrict__ bias,
                   float* __restrict__ C, int ldc, int colOff) {
    __shared__ unsigned short As[2][128 * LDS_STRIDE];

    const int tid  = threadIdx.x;
    const int wave = tid >> 5;
    const int lane = tid & 31;
    const int waveM = wave >> 2;
    const int waveN = wave & 3;
    const int rowBlock = blockIdx.x * 128;
    const int ktiles = Kd >> 5;

    v8f acc[4];
#pragma unroll
    for (int mi = 0; mi < 4; ++mi) acc[mi] = (v8f)0.0f;

    const int ldr    = tid >> 1;
    const int ldhalf = tid & 1;
    const size_t gOff = (size_t)(rowBlock + ldr) * 512 + ldhalf * 16;
    const int ldsOff = ldr * LDS_STRIDE + ldhalf * 16;
    const size_t nTile = (size_t)blockIdx.y * 4 + waveN;

    stage_async32(A1 + gOff, &As[0][ldsOff]);

    for (int kk = 0; kk < ktiles; ++kk) {
        const int cur = kk & 1;
        WAIT_ASYNC0();
        __syncthreads();

        if (kk + 1 < ktiles) {
            const int k1 = (kk + 1) * 32;
            const unsigned short* src =
                (k1 < KFEAT) ? (A1 + k1) : (A2 + (k1 - KFEAT));
            stage_async32(src + gOff, &As[1 - cur][ldsOff]);
        }

        v8u b = *(const v8u*)(Wp + (nTile * ktiles + kk) * 256 + (size_t)lane * 8);
        v16bf bfrag = __builtin_bit_cast(v16bf, b);

#pragma unroll
        for (int mi = 0; mi < 4; ++mi) {
            int rloc = waveM * 64 + mi * 16 + (lane & 15);
            int koff = (lane >> 4) * 16;
            const uint4* ap = (const uint4*)(&As[cur][rloc * LDS_STRIDE + koff]);
            v16bf afrag = as_bf16x16(ap[0], ap[1]);
            acc[mi] = wmma_bf16(afrag, bfrag, acc[mi]);
        }
    }

    const int col = blockIdx.y * 64 + waveN * 16 + (lane & 15);
    const float b = bias[col];
#pragma unroll
    for (int mi = 0; mi < 4; ++mi) {
        int row0 = rowBlock + waveM * 64 + mi * 16 + (lane >> 4) * 8;
#pragma unroll
        for (int e = 0; e < 8; ++e)
            C[(size_t)(row0 + e) * ldc + colOff + col] = acc[mi][e] + b;
    }
}

// ------------------------------ launch -------------------------------------

extern "C" void kernel_launch(void* const* d_in, const int* in_sizes, int n_in,
                              void* d_out, int out_size, void* d_ws, size_t ws_size,
                              hipStream_t stream) {
    (void)in_sizes; (void)n_in; (void)out_size; (void)ws_size;

    const float* inputs = (const float*)d_in[0];
    const float* Wf = (const float*)d_in[1];  const float* bf_ = (const float*)d_in[2];
    const float* Wi = (const float*)d_in[3];  const float* bi_ = (const float*)d_in[4];
    const float* Wo = (const float*)d_in[5];  const float* bo_ = (const float*)d_in[6];
    const float* Wc = (const float*)d_in[7];  const float* bc_ = (const float*)d_in[8];
    const float* Wl[4] = {(const float*)d_in[9],  (const float*)d_in[11],
                          (const float*)d_in[13], (const float*)d_in[15]};
    const float* bl[4] = {(const float*)d_in[10], (const float*)d_in[12],
                          (const float*)d_in[14], (const float*)d_in[16]};
    const float* Wg = (const float*)d_in[17]; const float* bg = (const float*)d_in[18];

    // workspace carve-up (256B aligned chunks)
    char* w = (char*)d_ws;
    size_t o = 0;
    auto carve = [&](size_t bytes) {
        void* p = w + o;
        o += (bytes + 255) & ~(size_t)255;
        return p;
    };
    unsigned short* Abf = (unsigned short*)carve((size_t)NROWS * KFEAT * 2);
    unsigned short* H0  = (unsigned short*)carve((size_t)NROWS * UNITS * 2);
    unsigned short* H1  = (unsigned short*)carve((size_t)NROWS * UNITS * 2);
    float*          cnd = (float*)carve((size_t)NROWS * UNITS * 4);
    unsigned*       WpG = (unsigned*)carve((size_t)KX * 2048 * 2);      // 4 gates packed
    unsigned*       WpL[4];
    const int NC[4] = {64, 128, 256, 512};
    for (int i = 0; i < 4; ++i)
        WpL[i] = (unsigned*)carve((size_t)UNITS * NC[i] * 2);
    unsigned*       WpWg = (unsigned*)carve((size_t)KX * NCSUM * 2);

    // --- prep: convert inputs, zero state, pack weights ---
    {
        int n = NROWS * KFEAT;
        k_f32_to_bf16<<<(n + 255) / 256, 256, 0, stream>>>(inputs, Abf, n);
    }
    hipMemsetAsync(H0,  0, (size_t)NROWS * UNITS * 2, stream);
    hipMemsetAsync(cnd, 0, (size_t)NROWS * UNITS * 4, stream);

    {
        int tot = (KX * UNITS) >> 1;                 // dwords per gate
        int blks = (tot + 255) / 256;
        k_pack_w<<<blks, 256, 0, stream>>>(Wf, WpG, KX, UNITS, 0);
        k_pack_w<<<blks, 256, 0, stream>>>(Wi, WpG, KX, UNITS, 32);
        k_pack_w<<<blks, 256, 0, stream>>>(Wo, WpG, KX, UNITS, 64);
        k_pack_w<<<blks, 256, 0, stream>>>(Wc, WpG, KX, UNITS, 96);
    }
    for (int i = 0; i < 4; ++i) {
        int tot = (UNITS * NC[i]) >> 1;
        k_pack_w<<<(tot + 255) / 256, 256, 0, stream>>>(Wl[i], WpL[i], UNITS, NC[i], 0);
    }
    {
        int tot = (KX * NCSUM) >> 1;
        k_pack_w<<<(tot + 255) / 256, 256, 0, stream>>>(Wg, WpWg, KX, NCSUM, 0);
    }

    // --- 4 recurrent layers + local logits ---
    float* outLocal  = (float*)d_out;                       // [N, 960]
    float* outGlobal = outLocal + (size_t)NROWS * NCSUM;    // [N, 960]
    unsigned short* Hbuf[2] = {H0, H1};
    const int colOff[4] = {0, 64, 192, 448};

    for (int l = 0; l < 4; ++l) {
        unsigned short* Hin  = Hbuf[l & 1];
        unsigned short* Hout = Hbuf[(l + 1) & 1];
        dim3 gc(NROWS / 128, UNITS / 64);
        k_lstm_cell<<<gc, 256, 0, stream>>>(Abf, Hin, Hout, cnd, WpG,
                                            bf_, bi_, bo_, bc_);
        dim3 gl(NROWS / 128, NC[l] / 64);
        k_gemm_logits<<<gl, 256, 0, stream>>>(Hout, nullptr, UNITS, WpL[l],
                                              bl[l], outLocal, NCSUM, colOff[l]);
    }

    // --- global logits: [inputs | h_final] @ Wg + bg ---
    dim3 gg(NROWS / 128, NCSUM / 64);
    k_gemm_logits<<<gg, 256, 0, stream>>>(Abf, Hbuf[0], KX, WpWg,
                                          bg, outGlobal, NCSUM, 0);
}
